// GraphUNet_7215545058055
// MI455X (gfx1250) — matrix operand order repaired and checked by reading, not compiled, not run on hardware
//
#include <hip/hip_runtime.h>
#include <hip/hip_bf16.h>
#include <math.h>

// ---------------- problem constants (from reference) ----------------
#define NODES   3072
#define EDGES   98304
#define INCH    128
#define CH      64          // per-head dim / hidden
#define HEADS   8
#define HC2     512         // HEADS*CH
#define KTOP    1536        // ceil(0.5*N)
#define NEG_BIG (-1e30f)

// ---------------- WMMA plumbing (CDNA5 gfx1250, wave32) ----------------
typedef __attribute__((ext_vector_type(16))) __bf16 bf16x16;
typedef __attribute__((ext_vector_type(8)))  float  f32x8;
typedef __attribute__((ext_vector_type(4)))  unsigned int u32x4;

union Frag {
    bf16x16 v;
    u32x4   q[2];
    __bf16  h[16];
};

__device__ __forceinline__ f32x8 zero8() {
    f32x8 z = {0.f, 0.f, 0.f, 0.f, 0.f, 0.f, 0.f, 0.f};
    return z;
}

__device__ __forceinline__ f32x8 wmma_bf16(bf16x16 a, bf16x16 b, f32x8 c) {
    // D = A(16x32 bf16) * B(32x16 bf16) + C(16x16 f32)
    return __builtin_amdgcn_wmma_f32_16x16x32_bf16(
        /*neg_a=*/false, a, /*neg_b=*/false, b,
        /*c_mod=*/(short)0, c, /*reuse_a=*/false, /*reuse_b=*/false);
}

// A-matrix fragment: 16x32 bf16 from row-major f32 source.
// lane: m = lane&15, half = lane>>4.
// VGPR i: K = half*16 + 2i
__device__ __forceinline__ bf16x16
load_a_frag_f32(const float* __restrict__ A, int lda, int row0, int kbase, int relu) {
    int lane = threadIdx.x & 31;
    int m = lane & 15, half = lane >> 4;
    const float* rp = A + (size_t)(row0 + m) * lda + kbase;
    bf16x16 f;
#pragma unroll
    for (int i = 0; i < 8; ++i) {
        int kk = half * 16 + 2 * i;
        float a0 = rp[kk], a1 = rp[kk + 1];
        if (relu) { a0 = fmaxf(a0, 0.f); a1 = fmaxf(a1, 0.f); }
        f[2 * i]     = (__bf16)a0;
        f[2 * i + 1] = (__bf16)a1;
    }
    return f;
}

// B-matrix fragment: 32x16 bf16 from row-major f32 [K,N] source.
// lane: n = lane&15, half = lane>>4; VGPR v: K = kbase + 16*half + 2v(+1)
__device__ __forceinline__ bf16x16
load_b_frag_f32(const float* __restrict__ B, int ldb, int kbase, int col0) {
    int lane = threadIdx.x & 31;
    int n = lane & 15, half = lane >> 4;
    bf16x16 f;
#pragma unroll
    for (int v = 0; v < 8; ++v) {
        int k = kbase + 16 * half + 2 * v;
        f[2 * v]     = (__bf16)B[(size_t)k * ldb + col0 + n];
        f[2 * v + 1] = (__bf16)B[(size_t)(k + 1) * ldb + col0 + n];
    }
    return f;
}

// B-matrix fragment where logical B[k][n] = S[col0+n][kbase+k]  (i.e. B = S^T)
__device__ __forceinline__ bf16x16
load_bT_frag_f32(const float* __restrict__ S, int lds_, int kbase, int col0) {
    int lane = threadIdx.x & 31;
    int n = lane & 15, half = lane >> 4;
    const float* p = S + (size_t)(col0 + n) * lds_ + kbase + 16 * half;
    bf16x16 f;
#pragma unroll
    for (int v = 0; v < 8; ++v) {
        f[2 * v]     = (__bf16)p[2 * v];
        f[2 * v + 1] = (__bf16)p[2 * v + 1];
    }
    return f;
}

// ---------------- LDS-blocked bf16-WMMA GEMM:  C = A@B (+bias) ----------------
// 128 threads (4 waves) per block; block computes a 64x64 output tile.
// Per 32-wide K step:
//   - A panel staged row-major As[m][k] (packed b128 stores),
//   - B panel staged TRANSPOSED Bs[n][k] so each lane's WMMA B-fragment is a
//     contiguous 32-byte run -> 2x ds_load_b128 instead of 16x ds_load_u16.
//   - A-fragment is two contiguous 16-byte runs -> 2x ds_load_b128.
// 80-byte row pitch keeps all chunks 16B-aligned and spreads banks.
// Requires M%64==0, N%64==0, K%32==0 (true for every GEMM in this network).
__global__ __launch_bounds__(128) void gemm_wmma_tiled(
    const float* __restrict__ A, const float* __restrict__ B,
    const float* __restrict__ bias, float* __restrict__ C,
    int Mm, int Nn, int Kk, int reluA) {
    __shared__ __align__(16) __bf16 As[64][40];   // As[m][k]
    __shared__ __align__(16) __bf16 Bs[64][40];   // Bs[n][k]  (transposed panel)

    int ntn = Nn >> 6;
    int tile = blockIdx.x;
    int tm = tile / ntn, tn = tile - tm * ntn;
    int row0 = tm << 6, col0 = tn << 6;

    int tid  = threadIdx.x;
    int wave = tid >> 5, lane = tid & 31;
    int n16 = lane & 15, half = lane >> 4;

    f32x8 acc[4];
#pragma unroll
    for (int j = 0; j < 4; ++j) acc[j] = zero8();

    for (int k0 = 0; k0 < Kk; k0 += 32) {
        // ---- stage A panel: 64 rows x 32 cols, 16 contiguous floats/thread,
        //      converted to bf16 and stored as two b128 chunks ----
        {
            int r = tid >> 1, c0 = (tid & 1) << 4;
            const float* src = A + (size_t)(row0 + r) * Kk + k0 + c0;
            Frag t;
#pragma unroll
            for (int e = 0; e < 16; ++e) {
                float a = src[e];
                if (reluA) a = fmaxf(a, 0.f);
                t.h[e] = (__bf16)a;
            }
            u32x4* d = (u32x4*)&As[r][c0];
            d[0] = t.q[0];
            d[1] = t.q[1];
        }
        // ---- stage B panel transposed: thread reads 16 contiguous floats of a
        //      B row (k fixed), scatters into Bs columns ----
        {
            int r = tid >> 2, c0 = (tid & 3) << 4;   // r = k-row 0..31, c0 = n
            const float* src = B + (size_t)(k0 + r) * Nn + col0 + c0;
#pragma unroll
            for (int e = 0; e < 16; ++e) Bs[c0 + e][r] = (__bf16)src[e];
        }
        __syncthreads();

        // ---- prefetch next K panels into cache (global_prefetch_b8) ----
        if (k0 + 32 < Kk) {
            __builtin_prefetch(A + (size_t)(row0 + (tid >> 1)) * Kk + k0 + 32 + ((tid & 1) << 4), 0, 1);
            __builtin_prefetch(B + (size_t)(k0 + 32 + (tid >> 2)) * Nn + col0 + ((tid & 3) << 4), 0, 1);
        }

        // ---- A fragment: two contiguous 16B runs (built once, reused 4x) ----
        Frag af;
        {
            const __bf16* row = &As[wave * 16 + n16][0];
            af.q[0] = *(const u32x4*)(row + half * 16);       // K = half*16 .. +7
            af.q[1] = *(const u32x4*)(row + half * 16 + 8);   // K = half*16+8 .. +15
        }
        // ---- 4 B fragments (each one contiguous 32B run) / 4 WMMAs ----
#pragma unroll
        for (int j = 0; j < 4; ++j) {
            Frag bf;
            const u32x4* pb = (const u32x4*)&Bs[j * 16 + n16][16 * half];
            bf.q[0] = pb[0];
            bf.q[1] = pb[1];
            acc[j] = wmma_bf16(af.v, bf.v, acc[j]);
        }
        __syncthreads();
    }
    // ---- epilogue: bias + store ----
#pragma unroll
    for (int j = 0; j < 4; ++j) {
        int cb = col0 + j * 16;
        float bv = bias ? bias[cb + n16] : 0.f;
#pragma unroll
        for (int i = 0; i < 8; ++i)
            C[(size_t)(row0 + wave * 16 + 8 * half + i) * Nn + cb + n16] = acc[j][i] + bv;
    }
}

// ---------------- dense adjacency scatter ----------------
__global__ void scatter_edges(float* __restrict__ A0, const int* __restrict__ ei,
                              const float* __restrict__ ea, int ne, int n) {
    int e = blockIdx.x * blockDim.x + threadIdx.x;
    if (e >= ne) return;
    int s = ei[e];
    int t = ei[ne + e];
    // edge weights are uniform in [0,1): atomicMax on uint bits is monotone
    // for non-negative floats -> deterministic duplicate resolution.
    atomicMax((unsigned int*)(A0 + (size_t)t * n + s), __float_as_uint(ea[e]));
}

// ---------------- g[t,h] = q_t . We_h ----------------
__global__ void compute_g(const float* __restrict__ q, const float* __restrict__ We,
                          float* __restrict__ g, int n) {
    int idx = blockIdx.x * blockDim.x + threadIdx.x;
    if (idx >= n * HEADS) return;
    int t = idx >> 3, h = idx & 7;
    float s = 0.f;
#pragma unroll 8
    for (int c = 0; c < CH; ++c)
        s += q[(size_t)t * HC2 + h * CH + c] * We[h * CH + c];
    g[idx] = s;
}

// ---------------- flash-style masked attention (one wave: 16 t-rows, 1 head) ----------------
// out[t, h*C+c] = sum_s alpha*v + (sum_s alpha*A)*We ;  stats (m,l) optionally spilled.
__global__ __launch_bounds__(32) void attn_flash(
    const float* __restrict__ q, const float* __restrict__ kmat,
    const float* __restrict__ vmat, const float* __restrict__ A,
    const float* __restrict__ g, const float* __restrict__ We,
    float* __restrict__ outp, float* __restrict__ ml, int n) {
    __shared__ __align__(16) float pls[16][36];

    int t0 = blockIdx.x << 4;
    int h  = blockIdx.y;
    int lane = threadIdx.x & 31, n16 = lane & 15, half = lane >> 4;

    // Q fragments for this tile (K-dim = C = 64 -> two 32-wide chunks)
    bf16x16 q0 = load_a_frag_f32(q + h * CH, HC2, t0, 0, 0);
    bf16x16 q1 = load_a_frag_f32(q + h * CH, HC2, t0, 32, 0);

    float grow[8], mrow[8], lrow[8], awrow[8];
#pragma unroll
    for (int i = 0; i < 8; ++i) {
        grow[i]  = g[(size_t)(t0 + 8 * half + i) * HEADS + h];
        mrow[i]  = NEG_BIG;
        lrow[i]  = 0.f;
        awrow[i] = 0.f;
    }
    f32x8 oacc[4];
#pragma unroll
    for (int j = 0; j < 4; ++j) oacc[j] = zero8();

    for (int s0 = 0; s0 < n; s0 += 32) {
        float sv[2][8], av[2][8];
#pragma unroll
        for (int sh = 0; sh < 2; ++sh) {
            bf16x16 b0 = load_bT_frag_f32(kmat + h * CH, HC2, 0,  s0 + sh * 16);
            bf16x16 b1 = load_bT_frag_f32(kmat + h * CH, HC2, 32, s0 + sh * 16);
            f32x8 lg = zero8();
            lg = wmma_bf16(q0, b0, lg);
            lg = wmma_bf16(q1, b1, lg);
#pragma unroll
            for (int i = 0; i < 8; ++i) {
                float a = A[(size_t)(t0 + 8 * half + i) * n + s0 + sh * 16 + n16];
                av[sh][i] = a;
                float s = (lg[i] + a * grow[i]) * 0.125f;   // 1/sqrt(64)
                sv[sh][i] = (a > 0.f) ? s : NEG_BIG;
            }
        }
        // online softmax: new per-row max (reduce over 16 lanes of same half)
        float scale[8];
#pragma unroll
        for (int i = 0; i < 8; ++i) {
            float x = fmaxf(sv[0][i], sv[1][i]);
#pragma unroll
            for (int msk = 8; msk >= 1; msk >>= 1)
                x = fmaxf(x, __shfl_xor(x, msk, 32));
            float newm = fmaxf(mrow[i], x);
            scale[i] = __expf(mrow[i] - newm);

            float p0 = __expf(sv[0][i] - newm); if (!(av[0][i] > 0.f)) p0 = 0.f;
            float p1 = __expf(sv[1][i] - newm); if (!(av[1][i] > 0.f)) p1 = 0.f;
            pls[8 * half + i][n16]      = p0;
            pls[8 * half + i][16 + n16] = p1;

            float ps = p0 + p1;
            float pa = p0 * av[0][i] + p1 * av[1][i];
#pragma unroll
            for (int msk = 8; msk >= 1; msk >>= 1) {
                ps += __shfl_xor(ps, msk, 32);
                pa += __shfl_xor(pa, msk, 32);
            }
            lrow[i]  = lrow[i]  * scale[i] + ps;
            awrow[i] = awrow[i] * scale[i] + pa;
            mrow[i]  = newm;
        }
#pragma unroll
        for (int j = 0; j < 4; ++j)
#pragma unroll
            for (int i = 0; i < 8; ++i) oacc[j][i] *= scale[i];

        __syncthreads();
        // re-read probabilities in A-fragment layout (LDS transpose of layouts)
        bf16x16 pf;
        {
            const float* row = &pls[n16][0];
#pragma unroll
            for (int i = 0; i < 8; ++i) {
                int kk = half * 16 + 2 * i;
                pf[2 * i]     = (__bf16)row[kk];
                pf[2 * i + 1] = (__bf16)row[kk + 1];
            }
        }
        __syncthreads();
        // out += P(16x32) @ V(32x64)
#pragma unroll
        for (int j = 0; j < 4; ++j) {
            bf16x16 vf = load_b_frag_f32(vmat + h * CH, HC2, s0, j * 16);
            oacc[j] = wmma_bf16(pf, vf, oacc[j]);
        }
    }
    // epilogue: normalize, add aw*We, write out; optionally spill (m,l)
#pragma unroll
    for (int i = 0; i < 8; ++i) {
        size_t row = t0 + 8 * half + i;
        float inv = 1.f / fmaxf(lrow[i], 1e-16f);
        float awn = awrow[i] * inv;
#pragma unroll
        for (int j = 0; j < 4; ++j) {
            int c = j * 16 + n16;
            outp[row * HC2 + h * CH + c] = oacc[j][i] * inv + awn * We[h * CH + c];
        }
        if (ml && n16 == 0) {
            ml[row * (2 * HEADS) + 2 * h]     = mrow[i];
            ml[row * (2 * HEADS) + 2 * h + 1] = lrow[i];
        }
    }
}

// ---------------- W0[t,s] = sum_h alpha[t,s,h]  (recompute logits via WMMA) ----------------
__global__ __launch_bounds__(32) void w0_alpha_sum(
    const float* __restrict__ q, const float* __restrict__ kmat,
    const float* __restrict__ A, const float* __restrict__ g,
    const float* __restrict__ ml, float* __restrict__ W0, int n) {
    int t0 = blockIdx.x << 4;
    int s0 = blockIdx.y << 5;
    int lane = threadIdx.x & 31, n16 = lane & 15, half = lane >> 4;

    float acc[2][8];
#pragma unroll
    for (int sh = 0; sh < 2; ++sh)
#pragma unroll
        for (int i = 0; i < 8; ++i) acc[sh][i] = 0.f;

    for (int h = 0; h < HEADS; ++h) {
        bf16x16 q0 = load_a_frag_f32(q + h * CH, HC2, t0, 0, 0);
        bf16x16 q1 = load_a_frag_f32(q + h * CH, HC2, t0, 32, 0);
#pragma unroll
        for (int sh = 0; sh < 2; ++sh) {
            bf16x16 b0 = load_bT_frag_f32(kmat + h * CH, HC2, 0,  s0 + sh * 16);
            bf16x16 b1 = load_bT_frag_f32(kmat + h * CH, HC2, 32, s0 + sh * 16);
            f32x8 lg = zero8();
            lg = wmma_bf16(q0, b0, lg);
            lg = wmma_bf16(q1, b1, lg);
#pragma unroll
            for (int i = 0; i < 8; ++i) {
                size_t r = t0 + 8 * half + i;
                float a  = A[r * n + s0 + sh * 16 + n16];
                float gv = g[r * HEADS + h];
                float m  = ml[r * (2 * HEADS) + 2 * h];
                float l  = ml[r * (2 * HEADS) + 2 * h + 1];
                float s  = (lg[i] + a * gv) * 0.125f;
                float e  = (a > 0.f) ? __expf(s - m) : 0.f;
                acc[sh][i] += e / fmaxf(l, 1e-16f);
            }
        }
    }
#pragma unroll
    for (int sh = 0; sh < 2; ++sh)
#pragma unroll
        for (int i = 0; i < 8; ++i)
            W0[(size_t)(t0 + 8 * half + i) * n + s0 + sh * 16 + n16] = acc[sh][i];
}

// ---------------- beta gate (in-place on out) ----------------
__global__ __launch_bounds__(32) void beta_combine(
    float* __restrict__ out, const float* __restrict__ xr,
    const float* __restrict__ wb) {
    int r = blockIdx.x, lane = threadIdx.x;
    float s = 0.f;
    for (int c = lane; c < HC2; c += 32) {
        float o = out[(size_t)r * HC2 + c], x = xr[(size_t)r * HC2 + c];
        s += o * wb[c] + x * wb[HC2 + c] + (o - x) * wb[2 * HC2 + c];
    }
#pragma unroll
    for (int msk = 16; msk >= 1; msk >>= 1) s += __shfl_xor(s, msk, 32);
    float b = 1.f / (1.f + __expf(-s));
    for (int c = lane; c < HC2; c += 32) {
        float o = out[(size_t)r * HC2 + c], x = xr[(size_t)r * HC2 + c];
        out[(size_t)r * HC2 + c] = b * x + (1.f - b) * o;
    }
}

// ---------------- elementwise / gather / scatter helpers ----------------
__global__ void add_store(const float* __restrict__ a, const float* __restrict__ b,
                          float* __restrict__ dst, size_t n) {
    size_t i = (size_t)blockIdx.x * blockDim.x + threadIdx.x;
    if (i < n) dst[i] = a[i] + b[i];
}

__global__ void copy_setdiag(const float* __restrict__ src, float* __restrict__ dst, int n) {
    size_t i = (size_t)blockIdx.x * blockDim.x + threadIdx.x;
    if (i >= (size_t)n * n) return;
    int r = (int)(i / n), c = (int)(i - (size_t)r * n);
    dst[i] = (r == c) ? 1.0f : src[i];
}

__global__ void pool_score(const float* __restrict__ h, const float* __restrict__ attn,
                           float* __restrict__ score, int n) {
    int i = blockIdx.x * blockDim.x + threadIdx.x;
    if (i >= n) return;
    float nrm = 0.f, d = 0.f;
#pragma unroll 8
    for (int c = 0; c < CH; ++c) nrm += attn[c] * attn[c];
    nrm = sqrtf(nrm);
#pragma unroll 8
    for (int c = 0; c < CH; ++c) d += h[(size_t)i * CH + c] * attn[c];
    score[i] = tanhf(d / nrm);
}

// exact descending top-k via rank counting (stable tie-break by index == lax.top_k)
__global__ void topk_rank(const float* __restrict__ score, int* __restrict__ perm,
                          float* __restrict__ vals, int n, int k) {
    int i = blockIdx.x * blockDim.x + threadIdx.x;
    if (i >= n) return;
    float si = score[i];
    int rank = 0;
    for (int j = 0; j < n; ++j) {
        float sj = score[j];
        rank += (sj > si) || (sj == si && j < i);
    }
    if (rank < k) { perm[rank] = i; vals[rank] = si; }
}

__global__ void gather_scale(const float* __restrict__ h, const int* __restrict__ perm,
                             const float* __restrict__ vals, float* __restrict__ hp, int k) {
    int idx = blockIdx.x * blockDim.x + threadIdx.x;
    int r = idx >> 6, c = idx & 63;
    if (r >= k) return;
    hp[idx] = h[(size_t)perm[r] * CH + c] * vals[r];
}

__global__ void gather_Ap(const float* __restrict__ A2, const int* __restrict__ perm,
                          float* __restrict__ Ap, int k, int n) {
    size_t idx = (size_t)blockIdx.x * blockDim.x + threadIdx.x;
    if (idx >= (size_t)k * k) return;
    int r = (int)(idx / k), c = (int)(idx - (size_t)r * k);
    Ap[idx] = A2[(size_t)perm[r] * n + perm[c]];
}

__global__ void scatter_add(float* __restrict__ hu, const int* __restrict__ perm,
                            const float* __restrict__ h2, int k) {
    int idx = blockIdx.x * blockDim.x + threadIdx.x;
    int r = idx >> 6, c = idx & 63;
    if (r >= k) return;
    hu[(size_t)perm[r] * CH + c] += h2[idx];   // perm rows unique -> no atomics
}

// ---------------- host-side helpers ----------------
struct TconvP {
    const float *Wq, *bq, *Wk, *bk, *Wv, *bv, *We, *Wskip, *bskip, *Wbeta;
};

static void run_gemm(const float* A, const float* B, const float* bias, float* C,
                     int M, int Nn, int Kk, int reluA, hipStream_t st) {
    dim3 grid((M / 64) * (Nn / 64));
    gemm_wmma_tiled<<<grid, 128, 0, st>>>(A, B, bias, C, M, Nn, Kk, reluA);
}

static void run_tconv(const float* xin, int n, int in_ch, const float* Aadj,
                      const TconvP& P, float* Q, float* Kb, float* Vb, float* XR,
                      float* G, float* ML, float* OUTP, float* final_dst,
                      bool beta, hipStream_t st) {
    run_gemm(xin, P.Wq,    P.bq,    Q,  n, HC2, in_ch, 0, st);
    run_gemm(xin, P.Wk,    P.bk,    Kb, n, HC2, in_ch, 0, st);
    run_gemm(xin, P.Wv,    P.bv,    Vb, n, HC2, in_ch, 0, st);
    run_gemm(xin, P.Wskip, P.bskip, XR, n, HC2, in_ch, 0, st);
    compute_g<<<(n * HEADS + 255) / 256, 256, 0, st>>>(Q, P.We, G, n);
    attn_flash<<<dim3(n / 16, HEADS), 32, 0, st>>>(Q, Kb, Vb, Aadj, G, P.We, OUTP, ML, n);
    if (beta) {
        beta_combine<<<n, 32, 0, st>>>(OUTP, XR, P.Wbeta);
    } else {
        size_t tot = (size_t)n * HC2;
        add_store<<<(unsigned)((tot + 255) / 256), 256, 0, st>>>(OUTP, XR, final_dst, tot);
    }
}

extern "C" void kernel_launch(void* const* d_in, const int* in_sizes, int n_in,
                              void* d_out, int out_size, void* d_ws, size_t ws_size,
                              hipStream_t stream) {
    (void)in_sizes; (void)n_in; (void)out_size; (void)ws_size;

    const float* x  = (const float*)d_in[0];
    const int*   ei = (const int*)d_in[1];
    const float* ea = (const float*)d_in[2];

    TconvP P0 = { (const float*)d_in[3],  (const float*)d_in[4],  (const float*)d_in[5],
                  (const float*)d_in[6],  (const float*)d_in[7],  (const float*)d_in[8],
                  (const float*)d_in[9],  (const float*)d_in[10], (const float*)d_in[11],
                  (const float*)d_in[12] };
    const float* lin0W = (const float*)d_in[13];
    const float* lin0b = (const float*)d_in[14];
    const float* pattn = (const float*)d_in[15];
    TconvP P1 = { (const float*)d_in[16], (const float*)d_in[17], (const float*)d_in[18],
                  (const float*)d_in[19], (const float*)d_in[20], (const float*)d_in[21],
                  (const float*)d_in[22], (const float*)d_in[23], (const float*)d_in[24],
                  (const float*)d_in[25] };
    const float* lin1W = (const float*)d_in[26];
    const float* lin1b = (const float*)d_in[27];
    TconvP PU = { (const float*)d_in[28], (const float*)d_in[29], (const float*)d_in[30],
                  (const float*)d_in[31], (const float*)d_in[32], (const float*)d_in[33],
                  (const float*)d_in[34], (const float*)d_in[35], (const float*)d_in[36],
                  nullptr };

    // -------- workspace carve-up (floats) --------
    float* w = (float*)d_ws;
    size_t o = 0;
    const size_t NN = (size_t)NODES * NODES;
    float* A0   = w + o; o += NN;                 // A0, later reused as diag-augmented A
    float* W0   = w + o; o += NN;
    float* A2   = w + o; o += NN;
    float* Q    = w + o; o += (size_t)NODES * HC2;
    float* Kb   = w + o; o += (size_t)NODES * HC2;
    float* Vb   = w + o; o += (size_t)NODES * HC2;
    float* XR   = w + o; o += (size_t)NODES * HC2;
    float* OUTP = w + o; o += (size_t)NODES * HC2;
    float* G    = w + o; o += (size_t)NODES * HEADS;
    float* ML   = w + o; o += (size_t)NODES * HEADS * 2;
    float* Hh   = w + o; o += (size_t)NODES * CH;
    float* HU   = w + o; o += (size_t)NODES * CH;
    float* SC   = w + o; o += NODES;
    float* VALS = w + o; o += KTOP;
    int*   PERM = (int*)(w + o); o += KTOP;
    float* HP   = w + o; o += (size_t)KTOP * CH;
    float* H2   = w + o; o += (size_t)KTOP * CH;
    float* AP   = w + o; o += (size_t)KTOP * KTOP;

    // -------- level-0 dense adjacency --------
    hipMemsetAsync(A0, 0, NN * sizeof(float), stream);
    scatter_edges<<<(EDGES + 255) / 256, 256, 0, stream>>>(A0, ei, ea, EDGES, NODES);

    // -------- down0 tconv (beta) + W0 --------
    run_tconv(x, NODES, INCH, A0, P0, Q, Kb, Vb, XR, G, ML, OUTP, nullptr, true, stream);
    w0_alpha_sum<<<dim3(NODES / 16, NODES / 32), 32, 0, stream>>>(Q, Kb, A0, G, ML, W0, NODES);

    // h = relu(h_conv) @ lin0.W + lin0.b
    run_gemm(OUTP, lin0W, lin0b, Hh, NODES, CH, HC2, 1, stream);

    // -------- augment: A = W0 w/ diag 1, A2 = A@A --------
    copy_setdiag<<<(unsigned)((NN + 255) / 256), 256, 0, stream>>>(W0, A0, NODES);
    run_gemm(A0, A0, nullptr, A2, NODES, NODES, NODES, 0, stream);

    // -------- TopK pooling --------
    pool_score<<<(NODES + 255) / 256, 256, 0, stream>>>(Hh, pattn, SC, NODES);
    topk_rank<<<(NODES + 255) / 256, 256, 0, stream>>>(SC, PERM, VALS, NODES, KTOP);
    gather_scale<<<(KTOP * CH + 255) / 256, 256, 0, stream>>>(Hh, PERM, VALS, HP, KTOP);
    gather_Ap<<<(unsigned)(((size_t)KTOP * KTOP + 255) / 256), 256, 0, stream>>>(A2, PERM, AP, KTOP, NODES);

    // -------- down1 tconv (beta), lin1 --------
    run_tconv(HP, KTOP, CH, AP, P1, Q, Kb, Vb, XR, G, nullptr, OUTP, nullptr, true, stream);
    run_gemm(OUTP, lin1W, lin1b, H2, KTOP, CH, HC2, 1, stream);

    // -------- unpool: hu = h ; hu[perm] += h2 --------
    hipMemcpyAsync(HU, Hh, (size_t)NODES * CH * sizeof(float),
                   hipMemcpyDeviceToDevice, stream);
    scatter_add<<<(KTOP * CH + 255) / 256, 256, 0, stream>>>(HU, PERM, H2, KTOP);

    // -------- up0 tconv on W0 graph (no beta): out = attn_out + skip --------
    run_tconv(HU, NODES, CH, W0, PU, Q, Kb, Vb, XR, G, nullptr, OUTP,
              (float*)d_out, false, stream);
}
